// OmniGAT_74208444940549
// MI455X (gfx1250) — compile-verified, matrix operand order checked by reading, not run on hardware
//
#include <hip/hip_runtime.h>
#include <hip/hip_bf16.h>

typedef __attribute__((ext_vector_type(16))) __bf16 v16bf;
typedef __attribute__((ext_vector_type(8)))  float  v8f;

#define LEAKY 0.2f
#define BNEPS 1e-5f

// ---------------------------------------------------------------- utilities
__global__ void k_zero_f32(float* p, long n) {
    long i = (long)blockIdx.x * blockDim.x + threadIdx.x;
    long stride = (long)gridDim.x * blockDim.x;
    for (; i < n; i += stride) p[i] = 0.f;
}

__global__ void k_cvt_bf16(const float* __restrict__ in, __bf16* __restrict__ out, long n) {
    long i = (long)blockIdx.x * blockDim.x + threadIdx.x;
    long stride = (long)gridDim.x * blockDim.x;
    for (; i < n; i += stride) out[i] = (__bf16)in[i];
}

// ---------------------------------------------------------------- weight pack
// Pack B[K,Nc] (f32 row-major) into WMMA B-operand order so each lane of a
// wave reads its full v16bf operand as one contiguous 32B load.
// Packed element index: ((kb32*tilesN + tn)*32 + lane)*16 + s
// maps to B[ kb32*32 + s + (s>=8)*8 + (lane>=16)*8 ][ tn*16 + (lane&15) ]
__global__ void k_pack_B(const float* __restrict__ B, __bf16* __restrict__ Bp,
                         int K, int Nc) {
    long i = (long)blockIdx.x * blockDim.x + threadIdx.x;
    long total = (long)K * Nc;
    if (i >= total) return;
    int  s    = (int)(i & 15);
    long t    = i >> 4;
    int  lane = (int)(t & 31);
    t >>= 5;
    int tilesN = Nc >> 4;
    int tn   = (int)(t % tilesN);
    int kb32 = (int)(t / tilesN);
    int col  = tn * 16 + (lane & 15);
    int kk   = kb32 * 32 + s + ((s >= 8) ? 8 : 0) + ((lane >> 4) & 1) * 8;
    Bp[i] = (__bf16)B[(size_t)kk * Nc + col];
}

// ---------------------------------------------------------------- WMMA GEMM
// D[M,Nc] f32 = A[M,K] bf16 (row major) x Bp (pre-packed WMMA operand order).
// One wave computes a 16x64 tile (4 accumulators); K fully unrolled.
// Operand packing per CDNA5 ISA 7.12.2:
//   A (16x32 bf16): lane<16 holds row M=lane&15, slot s -> K = s + (s>=8)*8,
//                   lanes>=16 same rows with K offset +8.
//   C/D (16x16 f32): VGPR r -> row = r + (lane>=16)*8, col = lane&15.
template <int K>
__global__ void k_gemm_bf16(const __bf16* __restrict__ A, const v16bf* __restrict__ Bp,
                            float* __restrict__ D, int M, int Nc) {
    constexpr int KB = K / 32;
    int w       = (blockIdx.x * blockDim.x + threadIdx.x) >> 5;
    int groupsN = Nc >> 6;                 // 64-column groups
    int tm      = w / groupsN;
    int tg      = w - tm * groupsN;
    if (tm * 16 >= M) return;
    int lane   = threadIdx.x & 31;
    int laneHi = (lane >> 4) & 1;
    int r16    = lane & 15;
    int tilesN = Nc >> 4;

    v8f acc0 = {}, acc1 = {}, acc2 = {}, acc3 = {};
    const __bf16* arow = A + (size_t)(tm * 16 + r16) * K;

#pragma unroll
    for (int kb32 = 0; kb32 < KB; ++kb32) {
        union { unsigned u[8]; v16bf v; } a;
#pragma unroll
        for (int v = 0; v < 8; ++v) {
            int kp = 2 * v + ((v >= 4) ? 8 : 0) + laneHi * 8;   // pair start K
            a.u[v] = *(const unsigned*)(arow + kb32 * 32 + kp); // 2 contiguous bf16
        }
        const v16bf* bb = Bp + ((size_t)kb32 * tilesN + tg * 4) * 32 + lane;
        v16bf b0 = bb[0];
        v16bf b1 = bb[32];
        v16bf b2 = bb[64];
        v16bf b3 = bb[96];
        acc0 = __builtin_amdgcn_wmma_f32_16x16x32_bf16(false, a.v, false, b0, (short)0, acc0, false, false);
        acc1 = __builtin_amdgcn_wmma_f32_16x16x32_bf16(false, a.v, false, b1, (short)0, acc1, false, false);
        acc2 = __builtin_amdgcn_wmma_f32_16x16x32_bf16(false, a.v, false, b2, (short)0, acc2, false, false);
        acc3 = __builtin_amdgcn_wmma_f32_16x16x32_bf16(false, a.v, false, b3, (short)0, acc3, false, false);
    }
#pragma unroll
    for (int r = 0; r < 8; ++r) {
        size_t row = (size_t)(tm * 16 + r + laneHi * 8);
        float* dp  = D + row * Nc + tg * 64 + r16;
        dp[0]  = acc0[r];
        dp[16] = acc1[r];
        dp[32] = acc2[r];
        dp[48] = acc3[r];
    }
}

// ---------------------------------------------------------------- edge pass 1
// One wave per edge. Lane covers 8 channels; head h = lane/8.
// e_h = sum_c leaky_relu(xl[src,h,c] + xr[dst,h,c]) * att[h,c]
// ex = exp(e_h) (max-subtraction skipped: softmax-invariant, logits are O(1))
// den[dst,h] += ex   (float atomic, L2-resident)
__global__ void k_edge_logits(const float* __restrict__ XL, const float* __restrict__ XR,
                              const float* __restrict__ att, const int* __restrict__ ei,
                              int E, int N, float* __restrict__ exb, float* __restrict__ den) {
    int t  = blockIdx.x * blockDim.x + threadIdx.x;
    int e  = t >> 5;
    int Ep = E + N;
    if (e >= Ep) return;
    int lane = threadIdx.x & 31;
    int s, d;
    if (e < E) { s = ei[e]; d = ei[E + e]; } else { s = d = e - E; }

    const float* xl = XL  + (size_t)s * 256 + lane * 8;
    const float* xr = XR  + (size_t)d * 256 + lane * 8;
    const float* at = att + lane * 8;
    float sum = 0.f;
#pragma unroll
    for (int j = 0; j < 8; ++j) {
        float v = xl[j] + xr[j];
        v = v > 0.f ? v : LEAKY * v;
        sum += v * at[j];
    }
    // reduce within each 8-lane head group
    sum += __shfl_xor(sum, 1, 32);
    sum += __shfl_xor(sum, 2, 32);
    sum += __shfl_xor(sum, 4, 32);
    float ex = expf(sum);
    if ((lane & 7) == 0) {
        int h = lane >> 3;
        exb[(size_t)e * 4 + h] = ex;
        atomicAdd(den + (size_t)d * 4 + h, ex);
    }
}

// ---------------------------------------------------------------- edge pass 2
// agg[dst,:] += alpha_h * xl[src,:]
__global__ void k_edge_aggr(const float* __restrict__ XL, const int* __restrict__ ei,
                            int E, int N, const float* __restrict__ exb,
                            const float* __restrict__ den, float* __restrict__ agg) {
    int t  = blockIdx.x * blockDim.x + threadIdx.x;
    int e  = t >> 5;
    int Ep = E + N;
    if (e >= Ep) return;
    int lane = threadIdx.x & 31;
    int s, d;
    if (e < E) { s = ei[e]; d = ei[E + e]; } else { s = d = e - E; }

    int h = lane >> 3;
    float alpha = exb[(size_t)e * 4 + h] / (den[(size_t)d * 4 + h] + 1e-16f);
    const float* xl = XL  + (size_t)s * 256 + lane * 8;
    float*       o  = agg + (size_t)d * 256 + lane * 8;
#pragma unroll
    for (int j = 0; j < 8; ++j) atomicAdd(o + j, alpha * xl[j]);
}

// ---------------------------------------------------------------- post layer 1
// h1 = elu(bn(agg + b1)) -> bf16 for layer-2 GEMM input
__global__ void k_post1(const float* __restrict__ agg, const float* __restrict__ b1,
                        const float* __restrict__ g1, const float* __restrict__ be1,
                        __bf16* __restrict__ h1b, long total) {
    long i = (long)blockIdx.x * blockDim.x + threadIdx.x;
    if (i >= total) return;
    int ch = (int)(i & 255);
    float inv = rsqrtf(1.f + BNEPS);
    float v = agg[i] + b1[ch];
    v = v * inv * g1[ch] + be1[ch];
    v = v > 0.f ? v : (expf(v) - 1.f);
    h1b[i] = (__bf16)v;
}

// ---------------------------------------------------------------- final fuse
// mean over heads -> +b2 -> bn -> elu -> dot(fcW)+fcb -> sigmoid
__global__ void k_final(const float* __restrict__ agg, const float* __restrict__ b2,
                        const float* __restrict__ g2, const float* __restrict__ be2,
                        const float* __restrict__ fcW, const float* __restrict__ fcb,
                        float* __restrict__ out, int N) {
    int i = blockIdx.x * blockDim.x + threadIdx.x;
    if (i >= N) return;
    const float* a = agg + (size_t)i * 256;
    float inv = rsqrtf(1.f + BNEPS);
    float s = 0.f;
#pragma unroll 4
    for (int c = 0; c < 64; ++c) {
        float v = 0.25f * (a[c] + a[64 + c] + a[128 + c] + a[192 + c]) + b2[c];
        v = v * inv * g2[c] + be2[c];
        v = v > 0.f ? v : (expf(v) - 1.f);
        s += v * fcW[c];
    }
    s += fcb[0];
    out[i] = 1.f / (1.f + expf(-s));
}

// ---------------------------------------------------------------- launcher
extern "C" void kernel_launch(void* const* d_in, const int* in_sizes, int n_in,
                              void* d_out, int out_size, void* d_ws, size_t ws_size,
                              hipStream_t stream) {
    const float* x    = (const float*)d_in[0];
    const int*   ei   = (const int*)  d_in[1];
    const float* W1l  = (const float*)d_in[2];
    const float* W1r  = (const float*)d_in[3];
    const float* att1 = (const float*)d_in[4];
    const float* b1   = (const float*)d_in[5];
    const float* g1   = (const float*)d_in[6];
    const float* be1  = (const float*)d_in[7];
    const float* W2l  = (const float*)d_in[8];
    const float* W2r  = (const float*)d_in[9];
    const float* att2 = (const float*)d_in[10];
    const float* b2   = (const float*)d_in[11];
    const float* g2   = (const float*)d_in[12];
    const float* be2  = (const float*)d_in[13];
    const float* fcW  = (const float*)d_in[14];
    const float* fcb  = (const float*)d_in[15];
    float* out = (float*)d_out;

    const int N  = in_sizes[0] / 128;   // 50000
    const int E  = in_sizes[1] / 2;     // 400000
    const int Ep = E + N;               // with self loops

    // workspace layout (256B aligned slices)
    char* ws = (char*)d_ws;
    size_t off = 0;
    auto take = [&](size_t bytes) -> char* {
        char* p = ws + off;
        off = (off + bytes + 255) & ~(size_t)255;
        return p;
    };
    __bf16* bx   = (__bf16*)take((size_t)N * 128 * 2);
    __bf16* bW1l = (__bf16*)take(128 * 256 * 2);
    __bf16* bW1r = (__bf16*)take(128 * 256 * 2);
    __bf16* bW2l = (__bf16*)take(256 * 256 * 2);
    __bf16* bW2r = (__bf16*)take(256 * 256 * 2);
    float*  XL   = (float*) take((size_t)N * 256 * 4);
    float*  XR   = (float*) take((size_t)N * 256 * 4);
    float*  exb  = (float*) take((size_t)Ep * 4 * 4);
    float*  den  = (float*) take((size_t)N * 4 * 4);
    float*  agg  = (float*) take((size_t)N * 256 * 4);
    __bf16* h1b  = (__bf16*)take((size_t)N * 256 * 2);
    (void)ws_size; (void)n_in; (void)out_size;

    const int BT = 256;
    auto blocks1d = [&](long threads) { return (unsigned)((threads + BT - 1) / BT); };

    // --- convert activations, pack weights into WMMA operand order
    k_cvt_bf16<<<blocks1d((long)N * 128), BT, 0, stream>>>(x, bx, (long)N * 128);
    k_pack_B<<<blocks1d(128 * 256), BT, 0, stream>>>(W1l, bW1l, 128, 256);
    k_pack_B<<<blocks1d(128 * 256), BT, 0, stream>>>(W1r, bW1r, 128, 256);
    k_pack_B<<<blocks1d(256 * 256), BT, 0, stream>>>(W2l, bW2l, 256, 256);
    k_pack_B<<<blocks1d(256 * 256), BT, 0, stream>>>(W2r, bW2r, 256, 256);

    // --- layer 1 GEMMs via WMMA: XL = x @ W1l, XR = x @ W1r  [N,256]
    long gwaves = (long)(N / 16) * (256 / 64);   // 16x64 tile per wave
    k_gemm_bf16<128><<<blocks1d(gwaves * 32), BT, 0, stream>>>(bx, (const v16bf*)bW1l, XL, N, 256);
    k_gemm_bf16<128><<<blocks1d(gwaves * 32), BT, 0, stream>>>(bx, (const v16bf*)bW1r, XR, N, 256);

    // --- layer 1 edge softmax + aggregation
    k_zero_f32<<<blocks1d((long)N * 4), BT, 0, stream>>>(den, (long)N * 4);
    k_zero_f32<<<blocks1d((long)N * 256), BT, 0, stream>>>(agg, (long)N * 256);
    k_edge_logits<<<blocks1d((long)Ep * 32), BT, 0, stream>>>(XL, XR, att1, ei, E, N, exb, den);
    k_edge_aggr  <<<blocks1d((long)Ep * 32), BT, 0, stream>>>(XL, ei, E, N, exb, den, agg);

    // --- BN + ELU -> bf16 hidden
    k_post1<<<blocks1d((long)N * 256), BT, 0, stream>>>(agg, b1, g1, be1, h1b, (long)N * 256);

    // --- layer 2 GEMMs (reuse XL/XR)
    k_gemm_bf16<256><<<blocks1d(gwaves * 32), BT, 0, stream>>>(h1b, (const v16bf*)bW2l, XL, N, 256);
    k_gemm_bf16<256><<<blocks1d(gwaves * 32), BT, 0, stream>>>(h1b, (const v16bf*)bW2r, XR, N, 256);

    // --- layer 2 edge softmax + aggregation (reuse den/exb/agg)
    k_zero_f32<<<blocks1d((long)N * 4), BT, 0, stream>>>(den, (long)N * 4);
    k_zero_f32<<<blocks1d((long)N * 256), BT, 0, stream>>>(agg, (long)N * 256);
    k_edge_logits<<<blocks1d((long)Ep * 32), BT, 0, stream>>>(XL, XR, att2, ei, E, N, exb, den);
    k_edge_aggr  <<<blocks1d((long)Ep * 32), BT, 0, stream>>>(XL, ei, E, N, exb, den, agg);

    // --- head-mean + BN + ELU + linear + sigmoid
    k_final<<<blocks1d(N), BT, 0, stream>>>(agg, b2, g2, be2, fcW, fcb, out, N);
}
